// SupConLoss_o_1726576853037
// MI455X (gfx1250) — matrix operand order repaired and checked by reading, compiled, not run on hardware
//
#include <hip/hip_runtime.h>
#include <math.h>

typedef __bf16 v16bf __attribute__((ext_vector_type(16)));
typedef __bf16 v8bf  __attribute__((ext_vector_type(8)));
typedef float  v8f   __attribute__((ext_vector_type(8)));

constexpr int WAVES  = 8;            // 8 waves * 16 rows = 128 rows per workgroup
constexpr int NT     = 4;            // column tiles per wave iteration (4 accumulators)
constexpr int NCHUNK = 8;            // column chunks (8192 / 8 = 1024 cols each)
constexpr int MAX_CHUNK_WORDS = 32;  // bitmask words, supports 1024 cols/chunk
#define TEMP_INV 14.285714285714285714f  /* 1 / 0.07 */
#define IGNORE_LBL (-100)

// ---------------------------------------------------------------------------
// Pass 0: split f32 features into bf16 hi + bf16 residual, in CONTRAST layout:
//   row c in [0,2*bsz): c<bsz -> features[c][0][:], else features[c-bsz][1][:]
// hi + lo reconstruction error ~2^-17 relative (residual capture is exact).
// ---------------------------------------------------------------------------
__global__ __launch_bounds__(256)
void supcon_convert(const float* __restrict__ features,
                    __bf16* __restrict__ chi, __bf16* __restrict__ clo,
                    int bsz, int d)
{
    size_t tid   = (size_t)blockIdx.x * blockDim.x + threadIdx.x;
    size_t total = (size_t)2 * bsz * d / 8;
    if (tid >= total) return;
    size_t e0 = tid * 8;                       // d % 8 == 0 -> never straddles rows
    int c = (int)(e0 / d);
    int k = (int)(e0 % d);
    int srow = (c < bsz) ? (c * 2) : ((c - bsz) * 2 + 1);
    const float* src = features + (size_t)srow * d + k;
    __bf16* ph = chi + (size_t)c * d + k;
    __bf16* pl = clo + (size_t)c * d + k;
#pragma unroll
    for (int i = 0; i < 8; ++i) {
        float x = src[i];
        __bf16 h = (__bf16)x;
        float r  = x - (float)h;               // exact (Sterbenz)
        ph[i] = h;
        pl[i] = (__bf16)r;
    }
}

// A-matrix 16x32 bf16 fragment: lane<16 holds K [0,8)+[16,24); base already +8*half
static __device__ __forceinline__ v16bf load_a_frag(const __bf16* p) {
    union { v16bf v; v8bf h[2]; } u;
    u.h[0] = *(const v8bf*)(p);
    u.h[1] = *(const v8bf*)(p + 16);
    return u.v;
}
// B-matrix 32x16 bf16 fragment: lane<16 holds K [0,8)+[16,24); base already +8*half
static __device__ __forceinline__ v16bf load_b_frag(const __bf16* p) {
    union { v16bf v; v8bf h[2]; } u;
    u.h[0] = *(const v8bf*)(p);
    u.h[1] = *(const v8bf*)(p + 16);
    return u.v;
}

// ---------------------------------------------------------------------------
// Pass 1: fused GEMM (bf16x3 split precision) + masked online softmax stats
// ---------------------------------------------------------------------------
__global__ __launch_bounds__(32 * WAVES)
void supcon_main(const __bf16* __restrict__ chi,     // (2*bsz, d) contrast hi
                 const __bf16* __restrict__ clo,     // (2*bsz, d) contrast lo
                 const int*    __restrict__ labels,  // (bsz)
                 const int*    __restrict__ topk,    // (bsz, kt)
                 float4*       __restrict__ ws4,     // (bsz, NCHUNK) {m,sumexp,sum_mask_s,pos}
                 int bsz, int d, int kt, int chunkCols)
{
    __shared__ int      lds_lab[WAVES * 16];
    __shared__ unsigned lds_bm[WAVES][16][MAX_CHUNK_WORDS];

    const int rg    = blockIdx.x / NCHUNK;
    const int chunk = blockIdx.x % NCHUNK;
    const int cbase = chunk * chunkCols;

    const int lane = threadIdx.x & 31;
    const int wave = threadIdx.x >> 5;
    const int half = lane >> 4;
    const int l15  = lane & 15;
    const int rbase = rg * (16 * WAVES) + wave * 16;

    // ---- build per-chunk top-k bitmask + row labels in LDS ----
    for (int i = threadIdx.x; i < WAVES * 16 * MAX_CHUNK_WORDS; i += blockDim.x)
        ((unsigned*)lds_bm)[i] = 0u;
    __syncthreads();
    if (lane < 16) {
        int row = rbase + lane;
        lds_lab[wave * 16 + lane] = labels[row];
        for (int j = 0; j < kt; ++j) {
            int t = topk[row * kt + j] - cbase;
            if (t >= 0 && t < chunkCols)
                lds_bm[wave][lane][t >> 5] |= (1u << (t & 31));
        }
    }
    __syncthreads();

    int li_v[8], row_v[8];
#pragma unroll
    for (int v = 0; v < 8; ++v) {
        int rl = v + 8 * half;                // C/D layout: VGPR v -> row v + 8*half
        li_v[v]  = lds_lab[wave * 16 + rl];
        row_v[v] = rbase + rl;
    }

    // anchor rows are contrast rows [0,bsz) (view 0)
    const __bf16* ah_p = chi + (size_t)(rbase + l15) * d + 8 * half;
    const __bf16* al_p = clo + (size_t)(rbase + l15) * d + 8 * half;

    float st_m[8], st_s[8], st_sms[8], st_pos[8];
#pragma unroll
    for (int v = 0; v < 8; ++v) { st_m[v] = -INFINITY; st_s[v] = 0.f; st_sms[v] = 0.f; st_pos[v] = 0.f; }

    for (int it = 0; it < chunkCols; it += 16 * NT) {
        int colv[NT];
        const __bf16* bh_p[NT];
        const __bf16* bl_p[NT];
        v8f acc[NT];
#pragma unroll
        for (int nt = 0; nt < NT; ++nt) {
            int c = cbase + it + nt * 16 + l15;
            colv[nt] = c;
            bh_p[nt] = chi + (size_t)c * d + 8 * half;
            bl_p[nt] = clo + (size_t)c * d + 8 * half;
            v8f z = {};
            acc[nt] = z;
        }
        // ---- K loop: bf16x3 (ah*bh + al*bh + ah*bl), K=32 per trio ----
#pragma unroll 2
        for (int kk = 0; kk < d; kk += 32) {
            v16bf ah = load_a_frag(ah_p + kk);
            v16bf al = load_a_frag(al_p + kk);
#pragma unroll
            for (int nt = 0; nt < NT; ++nt) {
                v16bf bh = load_b_frag(bh_p[nt] + kk);
                v16bf bl = load_b_frag(bl_p[nt] + kk);
                acc[nt] = __builtin_amdgcn_wmma_f32_16x16x32_bf16(
                    false, ah, false, bh, (short)0, acc[nt], false, false);
                acc[nt] = __builtin_amdgcn_wmma_f32_16x16x32_bf16(
                    false, al, false, bh, (short)0, acc[nt], false, false);
                acc[nt] = __builtin_amdgcn_wmma_f32_16x16x32_bf16(
                    false, ah, false, bl, (short)0, acc[nt], false, false);
            }
        }
        // ---- fused masked online-softmax epilogue ----
#pragma unroll
        for (int nt = 0; nt < NT; ++nt) {
            int c  = colv[nt];
            int cl = labels[(c < bsz) ? c : (c - bsz)];
            bool nz = (c < bsz) && (cl != 0);           // nz_cols (padded cols -> 0)
            int colc = it + nt * 16 + l15;              // column index within chunk
            unsigned bit = 1u << (colc & 31);
            int w = colc >> 5;
#pragma unroll
            for (int v = 0; v < 8; ++v) {
                int li = li_v[v];
                int rl = v + 8 * half;
                float s = acc[nt][v] * TEMP_INV;
                bool hit  = (lds_bm[wave][rl][w] & bit) != 0u;
                bool diag = (c == row_v[v]);
                bool ign  = (li == IGNORE_LBL);
                bool iso  = (li == 0);
                float maskv = (!ign && (iso ? hit : ((li == cl) && !diag))) ? 1.f : 0.f;
                float lmv   = (!ign && !diag && (iso ? (hit || nz) : true)) ? 1.f : 0.f;
                float mn = fmaxf(st_m[v], s);           // rowmax over ALL columns
                st_s[v]  = st_s[v] * __expf(st_m[v] - mn) + lmv * __expf(s - mn);
                st_m[v]  = mn;
                st_sms[v] = fmaf(maskv, s, st_sms[v]);
                st_pos[v] += maskv;
            }
        }
    }

    // ---- cross-lane reduce within each 16-lane half (same row set) ----
#pragma unroll
    for (int v = 0; v < 8; ++v) {
        for (int off = 1; off < 16; off <<= 1) {
            float m2 = __shfl_xor(st_m[v],  off, 32);
            float s2 = __shfl_xor(st_s[v],  off, 32);
            float a2 = __shfl_xor(st_sms[v], off, 32);
            float p2 = __shfl_xor(st_pos[v], off, 32);
            float mn = fmaxf(st_m[v], m2);
            st_s[v] = st_s[v] * __expf(st_m[v] - mn) + s2 * __expf(m2 - mn);
            st_m[v] = mn;
            st_sms[v] += a2;
            st_pos[v] += p2;
        }
    }
    if (l15 == 0) {
#pragma unroll
        for (int v = 0; v < 8; ++v) {
            float4 o; o.x = st_m[v]; o.y = st_s[v]; o.z = st_sms[v]; o.w = st_pos[v];
            ws4[(size_t)row_v[v] * NCHUNK + chunk] = o;
        }
    }
}

// ---------------------------------------------------------------------------
// Pass 2: merge chunks per row (log-sum-exp), reduce to scalar loss
// ---------------------------------------------------------------------------
__global__ __launch_bounds__(256)
void supcon_finalize(const float4* __restrict__ ws4,
                     const int*    __restrict__ labels,
                     float*        __restrict__ out, int bsz)
{
    __shared__ float sl[256], sc[256];
    float lsum = 0.f, cnt = 0.f;
    for (int r = threadIdx.x; r < bsz; r += 256) {
        float m = -INFINITY, S = 0.f, sms = 0.f, pos = 0.f;
        for (int ch = 0; ch < NCHUNK; ++ch) {
            float4 t = ws4[(size_t)r * NCHUNK + ch];
            float mn = fmaxf(m, t.x);
            S = S * __expf(m - mn) + t.y * __expf(t.x - mn);
            m = mn; sms += t.z; pos += t.w;
        }
        int li = labels[r];
        float val = pos * (float)(li + 1);
        bool valid = (val != 0.f);
        float denom = valid ? pos : 1.f;
        // sum(mask*log_prob) = sms - pos*m - pos*log(sum lm*exp(s-m))
        float mlpp = (sms - pos * m - pos * __logf(S)) / denom;
        float loss = -mlpp;                 // TEMP/BASE_TEMP == 1
        if (valid) { lsum += loss; cnt += 1.f; }
    }
    sl[threadIdx.x] = lsum; sc[threadIdx.x] = cnt;
    __syncthreads();
    for (int s = 128; s > 0; s >>= 1) {
        if (threadIdx.x < s) { sl[threadIdx.x] += sl[threadIdx.x + s];
                               sc[threadIdx.x] += sc[threadIdx.x + s]; }
        __syncthreads();
    }
    if (threadIdx.x == 0) out[0] = sl[0] / sc[0];
}

extern "C" void kernel_launch(void* const* d_in, const int* in_sizes, int n_in,
                              void* d_out, int out_size, void* d_ws, size_t ws_size,
                              hipStream_t stream)
{
    const float* features = (const float*)d_in[0];
    const int*   labels   = (const int*)d_in[1];
    const int*   topk     = (const int*)d_in[2];

    const int bsz = in_sizes[1];                 // 4096
    const int d   = in_sizes[0] / (bsz * 2);     // 768
    const int kt  = in_sizes[2] / bsz;           // 16
    const int ncol = bsz * 2;                    // 8192
    const int chunkCols = ncol / NCHUNK;         // 1024

    // workspace layout: [chi: 2*bsz*d bf16][clo: 2*bsz*d bf16][ws4: bsz*NCHUNK float4]
    const size_t nfeat = (size_t)ncol * d;
    __bf16* chi = (__bf16*)d_ws;
    __bf16* clo = chi + nfeat;
    float4* ws4 = (float4*)((char*)d_ws + 2 * nfeat * sizeof(__bf16));

    {
        size_t total = nfeat / 8;
        dim3 grid((unsigned)((total + 255) / 256));
        supcon_convert<<<grid, 256, 0, stream>>>(features, chi, clo, bsz, d);
    }
    dim3 grid((bsz / (16 * WAVES)) * NCHUNK);    // 32 row-groups * 8 chunks = 256
    supcon_main<<<grid, 32 * WAVES, 0, stream>>>(chi, clo, labels, topk, ws4,
                                                 bsz, d, kt, chunkCols);
    supcon_finalize<<<1, 256, 0, stream>>>(ws4, labels, (float*)d_out, bsz);
}
